// SparseLinear_9861244911617
// MI455X (gfx1250) — matrix operand order repaired and checked by reading, not compiled, not run on hardware
//
#include <hip/hip_runtime.h>

// ---------------------------------------------------------------------------
// SparseLinear on MI455X (gfx1250):
//   1) memset W (d_ws, 64 MB fp32)
//   2) COO scatter with global_atomic_add_f32
//   3) pack W fp32 -> {hi_bf16, lo_bf16} uint32, IN PLACE (bf16x3 split)
//   4) pack x  fp32 -> packed uint32 into d_ws + 64 MB
//   5) GEMM: y = x @ W^T + bias via v_wmma_f32_16x16x32_bf16
//      block 128x128, 8 wave32 (2Mx4N), wave tile 64x32 (4x2 WMMA tiles)
//      -> 24 WMMA per 24 ds_load_b128 per wave-step (vs 12:16 before)
// ---------------------------------------------------------------------------

#define IN_F  4096
#define OUT_F 4096

#define BM 128    // block tile M
#define BN 128    // block tile N
#define BK 32     // K step (== WMMA K)
#define MT 4      // 16-row M tiles per wave
#define NT 2      // 16-col N tiles per wave
#define LST 40    // LDS row stride (halves), padded vs bank conflicts
#define PLANE (BM * LST)          // halves per plane (X and W tiles both 128 rows)
#define STAGE (4 * PLANE)         // halves per stage: Xh | Xl | Wh | Wl
#define LDS_BYTES (2 * STAGE * 2) // two stages, 2 bytes per half = 81920

typedef __attribute__((ext_vector_type(16))) __bf16         v16bf;
typedef __attribute__((ext_vector_type(8)))  float          v8f;
typedef __attribute__((ext_vector_type(4)))  float          f32x4;
typedef __attribute__((ext_vector_type(2)))  unsigned int   u32x2;
typedef __attribute__((ext_vector_type(4)))  unsigned int   u32x4;
typedef __attribute__((ext_vector_type(8)))  unsigned short u16x8;
typedef __attribute__((ext_vector_type(16))) unsigned short u16x16;

__device__ __forceinline__ unsigned int rne_hi16(float f) {
    unsigned int u = __builtin_bit_cast(unsigned int, f);
    return u + 0x7FFFu + ((u >> 16) & 1u);     // bf16 RNE in bits [31:16]
}
// pack one fp32 into {hi_bf16[31:16], lo_bf16[15:0]} where f ~= hi + lo
__device__ __forceinline__ unsigned int pack_split(float f) {
    unsigned int hr = rne_hi16(f);
    float hf = __builtin_bit_cast(float, hr & 0xFFFF0000u);
    unsigned int lr = rne_hi16(f - hf);
    return (hr & 0xFFFF0000u) | (lr >> 16);
}

// ---------------------------------------------------------------------------
// Stage 2: densify COO weight into W[out][in]
// ---------------------------------------------------------------------------
__global__ void sl_scatter_kernel(const float* __restrict__ vals,
                                  const int* __restrict__ rows,
                                  const int* __restrict__ cols,
                                  float* __restrict__ W, int nnz) {
    int i = blockIdx.x * blockDim.x + threadIdx.x;
    if (i < nnz) {
        atomicAdd(&W[(size_t)rows[i] * IN_F + cols[i]], vals[i]);
    }
}

// ---------------------------------------------------------------------------
// Stage 3/4: fp32 -> packed {hi,lo} bf16 split (src may alias dst: in-place)
// ---------------------------------------------------------------------------
__global__ void sl_pack_kernel(const float* src, unsigned int* dst, int n4) {
    int i = blockIdx.x * blockDim.x + threadIdx.x;
    if (i < n4) {
        f32x4 v = ((const f32x4*)src)[i];
        u32x4 p;
#pragma unroll
        for (int e = 0; e < 4; ++e) p[e] = pack_split(v[e]);
        ((u32x4*)dst)[i] = p;
    }
}

// ---------------------------------------------------------------------------
// Stage 5: Y[m][n] = sum_k X[m][k] * W[n][k] + bias[n]
// ---------------------------------------------------------------------------
__global__ __launch_bounds__(256)
void sl_gemm_kernel(const unsigned int* __restrict__ xp,   // packed X
                    const unsigned int* __restrict__ wp,   // packed W
                    const float* __restrict__ bias,
                    float* __restrict__ out) {
    extern __shared__ __align__(16) unsigned short smem[]; // 2 stages x 4 planes

    const int tid   = threadIdx.x;
    const int lane  = tid & 31;          // wave32
    const int wave  = tid >> 5;          // 0..7
    const int waveM = wave >> 2;         // 0..1 -> 64 rows each
    const int waveN = wave & 3;          // 0..3 -> 32 cols each
    const int l     = lane & 15;
    const int h     = lane >> 4;         // lane half selects K sub-block

    const int m0 = blockIdx.y * BM;
    const int n0 = blockIdx.x * BN;

    // ---- global -> reg loaders (b128 of packed dwords) ----
    auto loadXreg = [&](int kk, u32x4* xr) {
#pragma unroll
        for (int j = 0; j < 4; ++j) {
            int i = tid + j * 256;                 // 1024 quads in a 128x32 tile
            int row = i >> 3, c4 = i & 7;
            xr[j] = *(const u32x4*)&xp[(size_t)(m0 + row) * IN_F + kk + c4 * 4];
        }
    };
    auto loadWreg = [&](int kk, u32x4* wr) {
#pragma unroll
        for (int j = 0; j < 4; ++j) {
            int i = tid + j * 256;
            int row = i >> 3, c4 = i & 7;
            wr[j] = *(const u32x4*)&wp[(size_t)(n0 + row) * IN_F + kk + c4 * 4];
        }
    };
    // ---- reg -> LDS: v_perm_b32 splits packed quads into planar hi/lo ----
    // hiPlane/loPlane are PLANE halves apart -> ds_store_2addr-friendly
    auto storeTile = [&](unsigned short* hiPlane, int i, u32x4 p) {
        int off = (i >> 3) * LST + (i & 7) * 4;    // halves
        u32x2 hi, lo;
        hi[0] = __builtin_amdgcn_perm(p[1], p[0], 0x07060302u);
        hi[1] = __builtin_amdgcn_perm(p[3], p[2], 0x07060302u);
        lo[0] = __builtin_amdgcn_perm(p[1], p[0], 0x05040100u);
        lo[1] = __builtin_amdgcn_perm(p[3], p[2], 0x05040100u);
        *(u32x2*)&hiPlane[off]         = hi;
        *(u32x2*)&hiPlane[off + PLANE] = lo;
    };
    // ---- LDS -> WMMA fragments (two ds_load_b128 each) ----
    // A (16x32, 16-bit): lane<16 holds K {0..7, 16..23}; lane>=16 {8..15, 24..31}
    auto fragA = [&](const unsigned short* base, int am) -> v16bf {
        u16x8 p0 = *(const u16x8*)&base[am * LST + h * 8];
        u16x8 p1 = *(const u16x8*)&base[am * LST + 16 + h * 8];
        u16x16 t;
#pragma unroll
        for (int e = 0; e < 8; ++e) { t[e] = p0[e]; t[8 + e] = p1[e]; }
        return __builtin_bit_cast(v16bf, t);
    };
    // B (32x16, 16-bit): lane<16 holds K {0..7, 16..23}; lane>=16 {8..15, 24..31}
    auto fragB = [&](const unsigned short* base, int bn) -> v16bf {
        u16x8 p0 = *(const u16x8*)&base[bn * LST + h * 8];
        u16x8 p1 = *(const u16x8*)&base[bn * LST + 16 + h * 8];
        u16x16 t;
#pragma unroll
        for (int e = 0; e < 8; ++e) { t[e] = p0[e]; t[8 + e] = p1[e]; }
        return __builtin_bit_cast(v16bf, t);
    };

    v8f acc[MT][NT];
#pragma unroll
    for (int mt = 0; mt < MT; ++mt)
#pragma unroll
        for (int nt = 0; nt < NT; ++nt) acc[mt][nt] = (v8f)(0.0f);

    u32x4 xr[4], wr[4];

    // prologue: fill stage 0  (layout per stage: Xh | Xl | Wh | Wl)
    loadXreg(0, xr);
    loadWreg(0, wr);
#pragma unroll
    for (int j = 0; j < 4; ++j) storeTile(smem,             tid + j * 256, xr[j]);
#pragma unroll
    for (int j = 0; j < 4; ++j) storeTile(smem + 2 * PLANE, tid + j * 256, wr[j]);
    __syncthreads();

    const int NK = IN_F / BK;
    for (int ks = 0; ks < NK; ++ks) {
        unsigned short* sb = smem + (ks & 1) * STAGE;        // current stage
        unsigned short* nb = smem + ((ks & 1) ^ 1) * STAGE;  // next stage
        const bool more = (ks + 1) < NK;
        if (more) {                       // issue next-stage global loads early
            loadXreg((ks + 1) * BK, xr);
            loadWreg((ks + 1) * BK, wr);
        }
        if (ks + 2 < NK) {                // L2/WGP$ prefetch of stage ks+2
            if (tid < BM)
                __builtin_prefetch(&xp[(size_t)(m0 + tid) * IN_F + (ks + 2) * BK], 0, 3);
            else
                __builtin_prefetch(&wp[(size_t)(n0 + tid - BM) * IN_F + (ks + 2) * BK], 0, 3);
        }

        // B fragments held across all mt iterations (reuse x4)
        v16bf bh[NT], bl[NT];
#pragma unroll
        for (int nt = 0; nt < NT; ++nt) {
            int bn = waveN * 32 + nt * 16 + l;
            bh[nt] = fragB(sb + 2 * PLANE, bn);
            bl[nt] = fragB(sb + 3 * PLANE, bn);
        }
#pragma unroll
        for (int mt = 0; mt < MT; ++mt) {
            int am = waveM * 64 + mt * 16 + l;
            v16bf ah = fragA(sb, am);
            v16bf al = fragA(sb + PLANE, am);
            // term-outer order: consecutive WMMAs hit different accumulators
#pragma unroll
            for (int nt = 0; nt < NT; ++nt)
                acc[mt][nt] = __builtin_amdgcn_wmma_f32_16x16x32_bf16(
                    false, ah, false, bh[nt], (short)0, acc[mt][nt], false, false);
#pragma unroll
            for (int nt = 0; nt < NT; ++nt)
                acc[mt][nt] = __builtin_amdgcn_wmma_f32_16x16x32_bf16(
                    false, ah, false, bl[nt], (short)0, acc[mt][nt], false, false);
#pragma unroll
            for (int nt = 0; nt < NT; ++nt)
                acc[mt][nt] = __builtin_amdgcn_wmma_f32_16x16x32_bf16(
                    false, al, false, bh[nt], (short)0, acc[mt][nt], false, false);
        }

        if (more) {                       // write other buffer; no race with cur
#pragma unroll
            for (int j = 0; j < 4; ++j) storeTile(nb,             tid + j * 256, xr[j]);
#pragma unroll
            for (int j = 0; j < 4; ++j) storeTile(nb + 2 * PLANE, tid + j * 256, wr[j]);
        }
        __syncthreads();
    }

    // epilogue: C/D layout -> VGPR r holds (M = r + 8*h, N = l); add bias
#pragma unroll
    for (int mt = 0; mt < MT; ++mt) {
        const int gm = m0 + waveM * 64 + mt * 16 + h * 8;
#pragma unroll
        for (int nt = 0; nt < NT; ++nt) {
            const int gn = n0 + waveN * 32 + nt * 16 + l;
            const float bv = bias[gn];
#pragma unroll
            for (int r = 0; r < 8; ++r) {
                out[(size_t)(gm + r) * OUT_F + gn] = acc[mt][nt][r] + bv;
            }
        }
    }
}

// ---------------------------------------------------------------------------
// Host side launcher (graph-capture safe: only async ops on `stream`)
// ---------------------------------------------------------------------------
extern "C" void kernel_launch(void* const* d_in, const int* in_sizes, int n_in,
                              void* d_out, int out_size, void* d_ws, size_t ws_size,
                              hipStream_t stream) {
    const float* x    = (const float*)d_in[0];
    const float* vals = (const float*)d_in[1];
    const int*   rows = (const int*)d_in[2];
    const int*   cols = (const int*)d_in[3];
    const float* bias = (const float*)d_in[4];
    float*       out  = (float*)d_out;

    float*        W   = (float*)d_ws;                       // 64 MB fp32 scratch
    unsigned int* Wp  = (unsigned int*)d_ws;                // packed in place
    unsigned int* Xp  = (unsigned int*)d_ws + (size_t)OUT_F * IN_F;  // +64 MB

    const int nnz  = in_sizes[1];
    const int nX   = in_sizes[0];          // 2*2048*4096
    const int mtot = nX / IN_F;            // 4096
    const int nW   = OUT_F * IN_F;

    // 1) zero dense W
    hipMemsetAsync(W, 0, (size_t)nW * sizeof(float), stream);

    // 2) COO scatter-add
    sl_scatter_kernel<<<(nnz + 255) / 256, 256, 0, stream>>>(vals, rows, cols, W, nnz);

    // 3) pack W in place (fp32 -> {hi,lo} bf16 pair per dword)
    sl_pack_kernel<<<(nW / 4 + 255) / 256, 256, 0, stream>>>(W, Wp, nW / 4);

    // 4) pack X into second scratch region
    sl_pack_kernel<<<(nX / 4 + 255) / 256, 256, 0, stream>>>(x, Xp, nX / 4);

    // 5) WMMA GEMM + bias (dynamic LDS: 2 stages x (Xh|Xl|Wh|Wl) = 80 KB)
    dim3 grid(OUT_F / BN, mtot / BM);      // (32, 32)
    sl_gemm_kernel<<<grid, 256, LDS_BYTES, stream>>>(Xp, Wp, bias, out);
}
